// CausalSelfAttention_50903952392467
// MI455X (gfx1250) — matrix operand order repaired and checked by reading, not compiled, not run on hardware
//
#include <hip/hip_runtime.h>

#define HEADS 12
#define HDIM  64
#define SEQ   1024
#define BATCH 8
#define CDIM  768
#define N3    2304

typedef __bf16 bf16_t;
typedef __attribute__((ext_vector_type(16))) __bf16 v16bf;
typedef __attribute__((ext_vector_type(8)))  __bf16 v8bf;
typedef __attribute__((ext_vector_type(8)))  float  v8f;

union Frag { v16bf v; v8bf h[2]; };

__device__ __forceinline__ bf16_t f2bf(float f) { return (bf16_t)f; }

__device__ __forceinline__ v8f vzero() {
  v8f z;
  #pragma unroll
  for (int i = 0; i < 8; ++i) z[i] = 0.0f;
  return z;
}

__device__ __forceinline__ v8f wmma_bf16(const Frag& a, const Frag& b, v8f c) {
  return __builtin_amdgcn_wmma_f32_16x16x32_bf16(false, a.v, false, b.v,
                                                 (short)0, c, false, false);
}

// ---- fast exp2 (v_exp_f32) ----
__device__ __forceinline__ float fexp2(float x) {
#if __has_builtin(__builtin_amdgcn_exp2f)
  return __builtin_amdgcn_exp2f(x);
#else
  return exp2f(x);
#endif
}

// ---- 16-lane butterfly max via v_permlane16_b32 (VALU, no LDS) ----
#if __has_builtin(__builtin_amdgcn_permlane16)
#define HAVE_PERMLANE16 1
__device__ __forceinline__ float permx(float x, unsigned s0, unsigned s1) {
  int i = __float_as_int(x);
  i = __builtin_amdgcn_permlane16(i, i, s0, s1, false, false);
  return __int_as_float(i);
}
#endif

__device__ __forceinline__ float rmax16(float x) {
#ifdef HAVE_PERMLANE16
  x = fmaxf(x, permx(x, 0x67452301u, 0xEFCDAB89u));  // lane ^ 1
  x = fmaxf(x, permx(x, 0x54761032u, 0xDCFE98BAu));  // lane ^ 2
  x = fmaxf(x, permx(x, 0x32107654u, 0xBA98FEDCu));  // lane ^ 4
  x = fmaxf(x, permx(x, 0xFEDCBA98u, 0x76543210u));  // lane ^ 8
#else
  x = fmaxf(x, __shfl_xor(x, 1, 16));
  x = fmaxf(x, __shfl_xor(x, 2, 16));
  x = fmaxf(x, __shfl_xor(x, 4, 16));
  x = fmaxf(x, __shfl_xor(x, 8, 16));
#endif
  return x;
}

// ---- CDNA5 async global->LDS copy (ASYNCcnt path), 16B per lane ----
__device__ __forceinline__ void async_b128(unsigned lds_off, unsigned voff,
                                           unsigned long long sbase) {
  asm volatile("global_load_async_to_lds_b128 %0, %1, %2"
               :: "v"(lds_off), "v"(voff), "s"(sbase) : "memory");
}
__device__ __forceinline__ void wait_async0() {
  asm volatile("s_wait_asynccnt 0" ::: "memory");
}

// ---------------- casts ----------------
__global__ void cast_f32_bf16(const float* __restrict__ src, bf16_t* __restrict__ dst, int n) {
  int i = blockIdx.x * blockDim.x + threadIdx.x;
  if (i < n) dst[i] = f2bf(src[i]);
}

// dst[n][k] = (bf16) src[k][n]   (src: [K rows, N cols] row-major)
__global__ void transpose_cast(const float* __restrict__ src, bf16_t* __restrict__ dst,
                               int K, int N) {
  int n = blockIdx.x * 16 + threadIdx.x;
  int k = blockIdx.y * 16 + threadIdx.y;
  if (n < N && k < K) dst[(size_t)n * K + k] = f2bf(src[(size_t)k * N + n]);
}

// ---------------- tiled WMMA GEMM: D = A[M,K] * Bt[N,K]^T + bias ----------------
// Double-buffered LDS filled by global_load_async_to_lds_b128.
// MODE 0: write f32 to outF ([M,768])
// MODE 1: QKV epilogue -> Qb/Kb [B,H,T,64] bf16 (Q scaled by log2e/8), Vt [B,H,64,T]
template <int MODE>
__global__ __launch_bounds__(256) void gemm_bf16_kernel(
    const bf16_t* __restrict__ A, const bf16_t* __restrict__ Bt,
    const float* __restrict__ bias, float* __restrict__ outF,
    bf16_t* __restrict__ Qb, bf16_t* __restrict__ Kb, bf16_t* __restrict__ Vt,
    int Kdim) {
  __shared__ __align__(16) bf16_t As[2][128][40];
  __shared__ __align__(16) bf16_t Bs[2][128][40];

  const int tid  = threadIdx.x;
  const int wave = tid >> 5, lane = tid & 31;
  const int hi = lane >> 4, lr = lane & 15;
  const int wm = wave & 3, wn = wave >> 2;
  const int m0 = blockIdx.y * 128, n0 = blockIdx.x * 128;
  const int koff8 = hi * 8, kb16 = hi * 16;

  // each thread owns 2 slots per tile; slot = 8 halfs (16B)
  const int s0 = tid, s1 = tid + 256;
  const int r0 = s0 >> 2, c0 = (s0 & 3) * 8;
  const int r1 = s1 >> 2, c1 = (s1 & 3) * 8;
  const unsigned long long aBase = (unsigned long long)(uintptr_t)A;
  const unsigned long long bBase = (unsigned long long)(uintptr_t)Bt;

  auto issue = [&](int ks, int buf) {
    const unsigned kk = (unsigned)(ks << 5);
    const unsigned va0 = ((unsigned)(m0 + r0) * (unsigned)Kdim + kk + c0) * 2u;
    const unsigned va1 = ((unsigned)(m0 + r1) * (unsigned)Kdim + kk + c1) * 2u;
    const unsigned vb0 = ((unsigned)(n0 + r0) * (unsigned)Kdim + kk + c0) * 2u;
    const unsigned vb1 = ((unsigned)(n0 + r1) * (unsigned)Kdim + kk + c1) * 2u;
    async_b128((unsigned)(uintptr_t)&As[buf][r0][c0], va0, aBase);
    async_b128((unsigned)(uintptr_t)&As[buf][r1][c1], va1, aBase);
    async_b128((unsigned)(uintptr_t)&Bs[buf][r0][c0], vb0, bBase);
    async_b128((unsigned)(uintptr_t)&Bs[buf][r1][c1], vb1, bBase);
  };

  v8f acc[2][4];
  #pragma unroll
  for (int i = 0; i < 2; ++i)
    #pragma unroll
    for (int j = 0; j < 4; ++j) acc[i][j] = vzero();

  const int ksteps = Kdim >> 5;
  issue(0, 0);
  for (int ks = 0; ks < ksteps; ++ks) {
    wait_async0();        // my async stores into buf (ks&1) have landed
    __syncthreads();      // everyone's landed; prev readers of other buf done
    if (ks + 1 < ksteps) issue(ks + 1, (ks + 1) & 1);  // prefetch next tile

    const int buf = ks & 1;
    Frag aF[2], bF[4];
    #pragma unroll
    for (int mt = 0; mt < 2; ++mt) {
      int r = wm * 32 + mt * 16 + lr;
      aF[mt].h[0] = *(const v8bf*)&As[buf][r][koff8];
      aF[mt].h[1] = *(const v8bf*)&As[buf][r][16 + koff8];
    }
    #pragma unroll
    for (int nt = 0; nt < 4; ++nt) {
      int c = wn * 64 + nt * 16 + lr;
      bF[nt].h[0] = *(const v8bf*)&Bs[buf][c][koff8];
      bF[nt].h[1] = *(const v8bf*)&Bs[buf][c][16 + koff8];
    }
    #pragma unroll
    for (int mt = 0; mt < 2; ++mt)
      #pragma unroll
      for (int nt = 0; nt < 4; ++nt)
        acc[mt][nt] = wmma_bf16(aF[mt], bF[nt], acc[mt][nt]);
  }

  // epilogue
  const float QSCALE = 0.125f * 1.44269504088896f;  // 1/sqrt(64) * log2(e)
  #pragma unroll
  for (int mt = 0; mt < 2; ++mt) {
    #pragma unroll
    for (int nt = 0; nt < 4; ++nt) {
      const int n = n0 + wn * 64 + nt * 16 + lr;
      const float bv = bias[n];
      #pragma unroll
      for (int r = 0; r < 8; ++r) {
        const int m = m0 + wm * 32 + mt * 16 + r + hi * 8;
        const float val = acc[mt][nt][r] + bv;
        if (MODE == 0) {
          outF[(size_t)m * CDIM + n] = val;
        } else {
          const int which = n / CDIM;
          const int cc = n - which * CDIM;
          const int h = cc >> 6, d = cc & 63;
          const int b = m >> 10, t = m & 1023;
          const size_t hd = (size_t)b * HEADS + h;
          if (which == 0)      Qb[(hd * SEQ + t) * HDIM + d] = f2bf(val * QSCALE);
          else if (which == 1) Kb[(hd * SEQ + t) * HDIM + d] = f2bf(val);
          else                 Vt[(hd * HDIM + d) * SEQ + t] = f2bf(val);
        }
      }
    }
  }
}

// ---------------- flash attention: one wave = 16 query rows ----------------
// Q pre-scaled by log2(e)/sqrt(Dh)  =>  softmax done in base-2 (v_exp_f32).
// Row-sum of P computed by an extra WMMA against an all-ones B matrix.
__global__ __launch_bounds__(256) void attn_kernel(
    const bf16_t* __restrict__ Qb, const bf16_t* __restrict__ Kb,
    const bf16_t* __restrict__ Vt, bf16_t* __restrict__ Ob) {
  __shared__ __align__(16) bf16_t Plds[8][16][40];

  const int tid  = threadIdx.x;
  const int wave = tid >> 5, lane = tid & 31;
  const int hi = lane >> 4, lr = lane & 15;
  const int koff8 = hi * 8, kb16 = hi * 16;
  const int bh = blockIdx.y;
  const int b = bh / HEADS, h = bh - b * HEADS;
  const int q0 = blockIdx.x * 128 + wave * 16;

  // Q fragments (A-matrix 16x32, two K-chunks over 64 features)
  const bf16_t* qp = Qb + ((size_t)bh * SEQ + q0) * HDIM + lr * HDIM;
  Frag aQ[2];
  aQ[0].h[0] = *(const v8bf*)(qp + koff8);
  aQ[0].h[1] = *(const v8bf*)(qp + 16 + koff8);
  aQ[1].h[0] = *(const v8bf*)(qp + 32 + koff8);
  aQ[1].h[1] = *(const v8bf*)(qp + 48 + koff8);

  // all-ones B fragment: D = P @ ones -> rowsum(P) replicated in every column
  Frag ones;
  #pragma unroll
  for (int i = 0; i < 16; ++i) ones.v[i] = (bf16_t)1.0f;

  float mrow[8], lrow[8];
  #pragma unroll
  for (int r = 0; r < 8; ++r) { mrow[r] = -1e30f; lrow[r] = 0.0f; }
  v8f oacc[4];
  #pragma unroll
  for (int nt = 0; nt < 4; ++nt) oacc[nt] = vzero();

  const bf16_t* kp = Kb + (size_t)bh * SEQ * HDIM;
  const bf16_t* vp = Vt + (size_t)bh * HDIM * SEQ;

  auto loadK = [&](Frag kf[2][2], int key0) {
    #pragma unroll
    for (int sn = 0; sn < 2; ++sn) {
      const bf16_t* kr = kp + (size_t)(key0 + sn * 16 + lr) * HDIM;
      kf[sn][0].h[0] = *(const v8bf*)(kr + koff8);
      kf[sn][0].h[1] = *(const v8bf*)(kr + 16 + koff8);
      kf[sn][1].h[0] = *(const v8bf*)(kr + 32 + koff8);
      kf[sn][1].h[1] = *(const v8bf*)(kr + 48 + koff8);
    }
  };

  Frag kc[2][2];
  loadK(kc, 0);

  const int kbmax = (q0 + 15) >> 5;
  for (int kb = 0; kb <= kbmax; ++kb) {
    const int key0 = kb * 32;

    // V loads first (consumed this iteration; waiting on them won't force
    // the later-issued K prefetch to have landed)
    Frag bV[4];
    #pragma unroll
    for (int nt = 0; nt < 4; ++nt) {
      const bf16_t* vr = vp + (size_t)(nt * 16 + lr) * SEQ + key0 + koff8;
      bV[nt].h[0] = *(const v8bf*)(vr);
      bV[nt].h[1] = *(const v8bf*)(vr + 16);
    }

    // S = Q * K^T  (two 16x16 column tiles, K-dim 64)
    v8f s[2];
    #pragma unroll
    for (int sn = 0; sn < 2; ++sn) {
      v8f t = vzero();
      t = wmma_bf16(aQ[0], kc[sn][0], t);
      t = wmma_bf16(aQ[1], kc[sn][1], t);
      s[sn] = t;
    }

    // prefetch next K block straight into kc (S WMMAs were the last readers)
    if (kb < kbmax) loadK(kc, key0 + 32);

    if (key0 + 31 > q0) {  // only diagonal block needs masking
      #pragma unroll
      for (int r = 0; r < 8; ++r) {
        const int rowg = q0 + r + hi * 8;
        if (key0 + lr      > rowg) s[0][r] = -1e30f;
        if (key0 + 16 + lr > rowg) s[1][r] = -1e30f;
      }
    }

    // online softmax (base-2); row-sum deferred to the ones-WMMA below
    float p0[8], p1[8], scl[8];
    #pragma unroll
    for (int r = 0; r < 8; ++r) {
      float mx = rmax16(fmaxf(s[0][r], s[1][r]));
      float mn = fmaxf(mrow[r], mx);
      scl[r] = fexp2(mrow[r] - mn);
      mrow[r] = mn;
      p0[r] = fexp2(s[0][r] - mn);
      p1[r] = fexp2(s[1][r] - mn);
    }
    #pragma unroll
    for (int nt = 0; nt < 4; ++nt)
      #pragma unroll
      for (int r = 0; r < 8; ++r) oacc[nt][r] *= scl[r];

    // P (D-layout f32) -> A-fragment (bf16) via per-wave LDS staging
    #pragma unroll
    for (int r = 0; r < 8; ++r) {
      const int row = r + hi * 8;
      Plds[wave][row][lr]      = f2bf(p0[r]);
      Plds[wave][row][lr + 16] = f2bf(p1[r]);
    }
    asm volatile("s_wait_dscnt 0" ::: "memory");
    Frag aP;
    aP.h[0] = *(const v8bf*)&Plds[wave][lr][koff8];
    aP.h[1] = *(const v8bf*)&Plds[wave][lr][16 + koff8];

    // rowsum(P) via matrix pipe; O += P * V
    v8f lsum = wmma_bf16(aP, ones, vzero());
    #pragma unroll
    for (int nt = 0; nt < 4; ++nt) oacc[nt] = wmma_bf16(aP, bV[nt], oacc[nt]);
    #pragma unroll
    for (int r = 0; r < 8; ++r) lrow[r] = lrow[r] * scl[r] + lsum[r];
  }

  // normalize and store context in [B,T,C] bf16
  bf16_t* op = Ob + ((size_t)b * SEQ + q0) * CDIM + h * HDIM;
  float inv[8];
  #pragma unroll
  for (int r = 0; r < 8; ++r) inv[r] = 1.0f / lrow[r];
  #pragma unroll
  for (int nt = 0; nt < 4; ++nt)
    #pragma unroll
    for (int r = 0; r < 8; ++r) {
      const int row = r + hi * 8;
      op[(size_t)row * CDIM + nt * 16 + lr] = f2bf(oacc[nt][r] * inv[r]);
    }
}

extern "C" void kernel_launch(void* const* d_in, const int* in_sizes, int n_in,
                              void* d_out, int out_size, void* d_ws, size_t ws_size,
                              hipStream_t stream) {
  (void)in_sizes; (void)n_in; (void)out_size; (void)ws_size;
  const float* x    = (const float*)d_in[0];
  const float* Wqkv = (const float*)d_in[1];
  const float* bqkv = (const float*)d_in[2];
  const float* Wout = (const float*)d_in[3];
  const float* bout = (const float*)d_in[4];
  float* out = (float*)d_out;

  char* ws = (char*)d_ws;
  size_t off = 0;
  auto carve = [&](size_t bytes) -> char* {
    char* p = ws + off;
    off += (bytes + 255) & ~(size_t)255;
    return p;
  };
  bf16_t* xb    = (bf16_t*)carve((size_t)BATCH * SEQ * CDIM * 2);
  bf16_t* wqkvT = (bf16_t*)carve((size_t)N3 * CDIM * 2);
  bf16_t* woutT = (bf16_t*)carve((size_t)CDIM * CDIM * 2);
  bf16_t* Qb    = (bf16_t*)carve((size_t)BATCH * HEADS * SEQ * HDIM * 2);
  bf16_t* Kbuf  = (bf16_t*)carve((size_t)BATCH * HEADS * SEQ * HDIM * 2);
  bf16_t* Vt    = (bf16_t*)carve((size_t)BATCH * HEADS * SEQ * HDIM * 2);
  bf16_t* Ob    = (bf16_t*)carve((size_t)BATCH * SEQ * CDIM * 2);

  const int nx = BATCH * SEQ * CDIM;
  cast_f32_bf16<<<nx / 256, 256, 0, stream>>>(x, xb, nx);
  transpose_cast<<<dim3(N3 / 16, CDIM / 16), dim3(16, 16), 0, stream>>>(Wqkv, wqkvT, CDIM, N3);
  transpose_cast<<<dim3(CDIM / 16, CDIM / 16), dim3(16, 16), 0, stream>>>(Wout, woutT, CDIM, CDIM);

  gemm_bf16_kernel<1><<<dim3(N3 / 128, (BATCH * SEQ) / 128), 256, 0, stream>>>(
      xb, wqkvT, bqkv, nullptr, Qb, Kbuf, Vt, CDIM);

  attn_kernel<<<dim3(SEQ / 128, BATCH * HEADS), 256, 0, stream>>>(Qb, Kbuf, Vt, Ob);

  gemm_bf16_kernel<0><<<dim3(CDIM / 128, (BATCH * SEQ) / 128), 256, 0, stream>>>(
      Ob, woutT, bout, out, nullptr, nullptr, nullptr, CDIM);
}
